// _CosineCodebook_82910048682286
// MI455X (gfx1250) — matrix operand order repaired and checked by reading, not compiled, not run on hardware
//
#include <hip/hip_runtime.h>
#include <math.h>

typedef __attribute__((ext_vector_type(16))) __bf16 v16bf;
typedef __attribute__((ext_vector_type(8)))  float  v8f;

// Problem constants (match reference)
#define NB 16384           // B rows (codes)
#define NC 1000            // classes
#define NK 4               // centroids per class
#define ND 64              // dim
#define NN (NC * NK)       // 4000 = GEMM N dimension
#define MTILES (NB / 16)   // 1024
#define NTILES (NN / 16)   // 250

#define CHUNK   50         // N-tiles staged in LDS per block (50*2KB = 100KB)
#define NCHUNKS (NTILES / CHUNK)  // 5
#define MB_PER_WAVE 2      // M-tiles per wave
#define WAVES   8          // waves per block
#define MB_PER_BLOCK (WAVES * MB_PER_WAVE)  // 16 M-tiles per block
#define MBLOCKS (MTILES / MB_PER_BLOCK)     // 64

// One-instruction cross-quad max: v_max_num_f32 with DPP quad_perm on src0.
// Stage 1: max with lane^1, stage 2: max with lane^2 (all lanes active).
__device__ __forceinline__ float quad_max_x1(float x) {
  float r;
  asm("v_max_num_f32_dpp %0, %1, %1 quad_perm:[1,0,3,2] row_mask:0xf bank_mask:0xf"
      : "=v"(r) : "v"(x));
  return r;
}
__device__ __forceinline__ float quad_max_x2(float x) {
  float r;
  asm("v_max_num_f32_dpp %0, %1, %1 quad_perm:[2,3,0,1] row_mask:0xf bank_mask:0xf"
      : "=v"(r) : "v"(x));
  return r;
}

// ---------------------------------------------------------------------------
// Pack codes (f32, row-major 16384x64) -> bf16 A fragments in WMMA VGPR order.
// A-frag (16x32 bf16): lanes 0-15 = rows 0-15 holding K = {0..7, 16..23},
//                      lanes 16-31 = rows 0-15 holding K = {8..15, 24..31}.
// Layout: ap[((mt*2 + s)*32 + lane)*16 + half], s = k-step (0,1).
// ---------------------------------------------------------------------------
__global__ __launch_bounds__(256) void pack_codes_kernel(
    const float* __restrict__ codes, __bf16* __restrict__ ap) {
  int t    = blockIdx.x * 256 + threadIdx.x;  // 65536 threads exactly
  int mt   = t >> 6;
  int rem  = t & 63;
  int s    = rem >> 5;
  int lane = rem & 31;
  int ml   = lane & 15;
  int hi   = lane >> 4;
  const float* row = codes + (size_t)(mt * 16 + ml) * ND + s * 32;
  v16bf v;
#pragma unroll
  for (int h = 0; h < 16; ++h) {
    int dk;
    if (hi == 0) dk = (h < 8) ? h      : h + 8;   // K 0..7, 16..23
    else         dk = (h < 8) ? h + 8  : h + 16;  // K 8..15, 24..31
    v[h] = (__bf16)row[dk];
  }
  *(v16bf*)(ap + (size_t)t * 16) = v;
}

// ---------------------------------------------------------------------------
// Normalize centroids (f32, (C*K) x 64) and pack -> bf16 B fragments.
// B-frag (32x16 bf16): lane n (0-15) holds column n, K = {0..7, 16..23};
//                      lanes 16-31 hold same columns, K = {8..15, 24..31}.
// Layout: bp[((nt*2 + s)*32 + lane)*16 + half].
// ---------------------------------------------------------------------------
__global__ __launch_bounds__(256) void pack_cents_kernel(
    const float* __restrict__ cents, __bf16* __restrict__ bp) {
  int t = blockIdx.x * 256 + threadIdx.x;
  if (t >= NTILES * 64) return;  // 16000 threads
  int nt   = t >> 6;
  int rem  = t & 63;
  int s    = rem >> 5;
  int lane = rem & 31;
  int nl   = lane & 15;
  int hi   = lane >> 4;
  const float* row = cents + (size_t)(nt * 16 + nl) * ND;
  float ss = 0.f;
#pragma unroll
  for (int d = 0; d < ND; ++d) { float x = row[d]; ss += x * x; }
  float rn = 1.0f / fmaxf(sqrtf(ss), 1e-12f);
  v16bf v;
#pragma unroll
  for (int h = 0; h < 16; ++h) {
    int d = s * 32 + ((h < 8) ? hi * 8 + h : 16 + hi * 8 + (h - 8));
    v[h] = (__bf16)(row[d] * rn);
  }
  *(v16bf*)(bp + (size_t)t * 16) = v;
}

// ---------------------------------------------------------------------------
// GEMM + min-over-4. Block = 8 waves, each wave owns 2 M-tiles; A fragments
// live in registers for the whole N sweep. Each block stages a 50-tile chunk
// of packed B (100 KB) into LDS once, then all waves stream B frags from LDS.
// Per N-tile: 4x v_wmma_f32_16x16x32_bf16, fused DPP quad-max epilogue,
// predicated stores (quad leaders).
// ---------------------------------------------------------------------------
__global__ __launch_bounds__(256) void gemm_min_kernel(
    const __bf16* __restrict__ ap, const __bf16* __restrict__ bp,
    float* __restrict__ out) {
  __shared__ __align__(16) __bf16 bsh[CHUNK * 1024];  // 100 KB

  const int tid   = threadIdx.x;
  const int lane  = tid & 31;
  const int wave  = tid >> 5;
  const int chunk = blockIdx.x % NCHUNKS;             // which 50 N-tiles
  const int mblk  = blockIdx.x / NCHUNKS;             // which 16 M-tiles
  const int mt0   = mblk * MB_PER_BLOCK + wave * MB_PER_WAVE;

  // --- A fragments: loaded once, kept in VGPRs for all 50 N-tiles ---
  v16bf a[MB_PER_WAVE][2];
#pragma unroll
  for (int mi = 0; mi < MB_PER_WAVE; ++mi)
#pragma unroll
    for (int s = 0; s < 2; ++s)
      a[mi][s] = *(const v16bf*)(ap +
          (((size_t)(mt0 + mi) * 2 + s) * 32 + lane) * 16);

  // --- Stage 50 B tiles (100 KB) into LDS cooperatively ---
  {
    const uint4* src = (const uint4*)(bp + (size_t)chunk * CHUNK * 1024);
    uint4* dst = (uint4*)bsh;
#pragma unroll
    for (int i = tid; i < CHUNK * 1024 * 2 / 16; i += 256)  // 6400 uint4
      dst[i] = src[i];
  }
  __syncthreads();

  // --- Sweep the chunk's N-tiles ---
  for (int t = 0; t < CHUNK; ++t) {
    const int nt = chunk * CHUNK + t;
    const v16bf* Bf = (const v16bf*)(bsh + (size_t)t * 1024);
    v16bf b0 = Bf[lane];        // k-step 0
    v16bf b1 = Bf[32 + lane];   // k-step 1

#pragma unroll
    for (int mi = 0; mi < MB_PER_WAVE; ++mi) {
      v8f acc = {};
      acc = __builtin_amdgcn_wmma_f32_16x16x32_bf16(false, a[mi][0], false, b0,
                                                    (short)0, acc, false, false);
      acc = __builtin_amdgcn_wmma_f32_16x16x32_bf16(false, a[mi][1], false, b1,
                                                    (short)0, acc, false, false);
      float res[8];
#pragma unroll
      for (int v = 0; v < 8; ++v) {
        float x = acc[v];
        x = quad_max_x1(x);         // max with lane^1 (single v_max_num_f32_dpp)
        x = quad_max_x2(x);         // max with lane^2
        res[v] = 1.0f - x;          // min(1 - sim) == 1 - max(sim)
      }
      if ((lane & 3) == 0) {        // quad leaders: classes nt*4 + quad
        int hi   = lane >> 4;
        int ccol = nt * 4 + ((lane & 15) >> 2);
        size_t base = (size_t)((mt0 + mi) * 16 + hi * 8) * NC + ccol;
#pragma unroll
        for (int v = 0; v < 8; ++v) out[base + (size_t)v * NC] = res[v];
      }
    }
  }
}

extern "C" void kernel_launch(void* const* d_in, const int* in_sizes, int n_in,
                              void* d_out, int out_size, void* d_ws, size_t ws_size,
                              hipStream_t stream) {
  const float* codes = (const float*)d_in[0];  // (16384, 64) f32
  const float* cents = (const float*)d_in[1];  // (1000, 4, 64) f32
  float* out = (float*)d_out;                  // (16384, 1000) f32

  __bf16* ap = (__bf16*)d_ws;                                 // 2 MB
  __bf16* bp = (__bf16*)((char*)d_ws + (size_t)NB * ND * 2);  // +512 KB

  pack_codes_kernel<<<(NB * 4) / 256, 256, 0, stream>>>(codes, ap);
  pack_cents_kernel<<<(NTILES * 64 + 255) / 256, 256, 0, stream>>>(cents, bp);
  gemm_min_kernel<<<MBLOCKS * NCHUNKS, 256, 0, stream>>>(ap, bp, out);
}
